// MultiHeadLinearAttention_15375982919767
// MI455X (gfx1250) — matrix operand order repaired and checked by reading, not compiled, not run on hardware
//
#include <hip/hip_runtime.h>

// ---------------------------------------------------------------------------
// MultiHeadLinearAttention for MI455X (gfx1250, wave32, WMMA).
// Pipeline:
//   1) f32 -> bf16 conversion of x and the four weight matrices (one pass).
//   2) Fused QKV GEMM (bf16 WMMA, f32 accum, double-buffered LDS), epilogue
//      fuses bias + elu()+1 for Q/K, writes bf16 activations.
//   3) Chunked linear-attention scan: all per-chunk math expressed as
//      64x64x64 bf16 WMMA GEMMs; f32 state in LDS.
//   4) Output projection GEMM (bf16 WMMA), f32 result to d_out.
// ---------------------------------------------------------------------------

typedef __attribute__((ext_vector_type(16))) __bf16 v16bf;
typedef __attribute__((ext_vector_type(8)))  float  v8f;

union FragBF { v16bf v; unsigned int u[8]; };

constexpr int B_   = 4;
constexpr int S_   = 2048;
constexpr int D_   = 1024;
constexpr int H_   = 16;
constexpr int HD_  = 64;
constexpr int MTOT = B_ * S_;   // 8192 rows

// ---- bf16 helpers (bit-level, round-to-nearest-even) ----------------------
__device__ __forceinline__ unsigned short f2bf(float f) {
    unsigned int u = __float_as_uint(f);
    unsigned int r = (u + 0x7FFFu + ((u >> 16) & 1u)) >> 16;
    return (unsigned short)r;
}
__device__ __forceinline__ float bf2f(unsigned short s) {
    return __uint_as_float(((unsigned int)s) << 16);
}

// ---- fragment loads from K-contiguous LDS tiles ---------------------------
// A operand: matrix stored [M][K] bf16, K contiguous. 16-bit A layout:
//   lane half h, row m = lane&15 (+base); vgpr i holds K pair
//   kb = 2*i + (i>=4 ? 8 : 0) + 8*h  (relative to kk).
__device__ __forceinline__ void load_frag_a(FragBF& f, const unsigned int* baseDw,
                                            int row, int kk, int strideDw, int half) {
#pragma unroll
    for (int i = 0; i < 8; ++i) {
        int kb = kk + 2 * i + ((i >= 4) ? 8 : 0) + 8 * half;
        f.u[i] = baseDw[row * strideDw + (kb >> 1)];
    }
}
// B operand: matrix stored [N][K] bf16 (K contiguous). Lane = N (&15),
// vgpr v holds K pair kb = 2*v + (v>=4 ? 8 : 0) + 8*h (relative to kk).
__device__ __forceinline__ void load_frag_b(FragBF& f, const unsigned int* baseDw,
                                            int row, int kk, int strideDw, int half) {
#pragma unroll
    for (int i = 0; i < 8; ++i) {
        int kb = kk + 2 * i + ((i >= 4) ? 8 : 0) + 8 * half;
        f.u[i] = baseDw[row * strideDw + (kb >> 1)];
    }
}

// 64x64x64 GEMM accumulate: acc += A(64x64, [M][K]) * B([N][K]).
// One wave computes a 32x32 tile selected by (wm, wn). Stride = 32 dwords.
__device__ __forceinline__ void mma64(const unsigned int* Adw, const unsigned int* Bdw,
                                      int wm, int wn, int half, int lr, v8f acc[2][2]) {
#pragma unroll
    for (int kk = 0; kk < 64; kk += 32) {
        FragBF a[2], b[2];
#pragma unroll
        for (int f = 0; f < 2; ++f) {
            load_frag_a(a[f], Adw, wm * 32 + f * 16 + lr, kk, 32, half);
            load_frag_b(b[f], Bdw, wn * 32 + f * 16 + lr, kk, 32, half);
        }
#pragma unroll
        for (int fm = 0; fm < 2; ++fm)
#pragma unroll
            for (int fn = 0; fn < 2; ++fn)
                acc[fm][fn] = __builtin_amdgcn_wmma_f32_16x16x32_bf16(
                    false, a[fm].v, false, b[fn].v, (short)0, acc[fm][fn], false, false);
    }
}

// ---------------------------------------------------------------------------
// f32 -> bf16 pack (2 elements / thread)
// ---------------------------------------------------------------------------
__global__ void cvt_f32_bf16_kernel(const float* __restrict__ src,
                                    unsigned short* __restrict__ dst, int n) {
    int i = (blockIdx.x * blockDim.x + threadIdx.x) * 2;
    if (i + 1 < n) {
        float2 f = *(const float2*)(src + i);
        unsigned int packed = (unsigned int)f2bf(f.x) | ((unsigned int)f2bf(f.y) << 16);
        *(unsigned int*)(dst + i) = packed;
    }
}

// ---------------------------------------------------------------------------
// Tiled bf16 GEMM: C(MTOT x D) = A(MTOT x D) * W(D x D) + bias
// Block tile 128x128, BK=32, 256 threads = 8 waves in a 4x2 wave grid,
// each wave a 32x64 tile (2x4 WMMA fragments). Double-buffered LDS:
// next tile's global loads are issued before the WMMA block; the
// s_wait_loadcnt lands at the ds_store after the math.
// ---------------------------------------------------------------------------
constexpr int TILE_SH = 128 * 32;  // shorts per LDS tile (A and B each)

__device__ __forceinline__ void gemm_body(const unsigned short* __restrict__ A,
                                          const unsigned short* __restrict__ W,
                                          const float* __restrict__ bias,
                                          unsigned short* __restrict__ obf,
                                          float* __restrict__ of32,
                                          bool act,
                                          unsigned short* As, unsigned short* Bs) {
    const int tid  = threadIdx.x;
    const int m0   = blockIdx.y * 128;
    const int n0   = blockIdx.x * 128;
    const int wave = tid >> 5;
    const int wm   = wave >> 1, wn = wave & 1;   // 4 x 2 wave grid
    const int lane = tid & 31, half = lane >> 4, lr = lane & 15;

    // Per-thread staging coordinates (two 16B chunks per tile each).
    int rowA[2], cbA[2], kB[2], nbB[2];
#pragma unroll
    for (int it = 0; it < 2; ++it) {
        int ca = tid + it * 256;          // A: 128 rows x 32 K
        rowA[it] = ca >> 2;
        cbA[it]  = (ca & 3) * 8;
        int cb = tid + it * 256;          // B: 32 K x 128 N
        kB[it]  = cb >> 4;
        nbB[it] = (cb & 15) * 8;
    }

    uint4 ra[2], rb[2];
    auto load_tiles = [&](int k0) {
#pragma unroll
        for (int it = 0; it < 2; ++it)
            ra[it] = *(const uint4*)(A + (m0 + rowA[it]) * D_ + k0 + cbA[it]);
#pragma unroll
        for (int it = 0; it < 2; ++it)
            rb[it] = *(const uint4*)(W + (k0 + kB[it]) * D_ + n0 + nbB[it]);
        if (k0 + 32 < D_) {   // L2 prefetch hint for the tile after next
            __builtin_prefetch(A + (m0 + rowA[0]) * D_ + k0 + 32, 0, 1);
            __builtin_prefetch(W + (k0 + 32 + kB[0]) * D_ + n0, 0, 1);
        }
    };
    auto store_tiles = [&](int buf) {
        unsigned short* as = As + buf * TILE_SH;
        unsigned short* bs = Bs + buf * TILE_SH;
#pragma unroll
        for (int it = 0; it < 2; ++it)
            *(uint4*)(as + rowA[it] * 32 + cbA[it]) = ra[it];
#pragma unroll
        for (int it = 0; it < 2; ++it) {
            unsigned int parts[4] = {rb[it].x, rb[it].y, rb[it].z, rb[it].w};
#pragma unroll
            for (int p = 0; p < 4; ++p) {     // transposed store: Bs[n][k]
                bs[(nbB[it] + 2 * p)     * 32 + kB[it]] = (unsigned short)(parts[p] & 0xffffu);
                bs[(nbB[it] + 2 * p + 1) * 32 + kB[it]] = (unsigned short)(parts[p] >> 16);
            }
        }
    };

    const v8f zero8 = {0.f, 0.f, 0.f, 0.f, 0.f, 0.f, 0.f, 0.f};
    v8f acc[2][4];
#pragma unroll
    for (int fm = 0; fm < 2; ++fm)
#pragma unroll
        for (int fn = 0; fn < 4; ++fn) acc[fm][fn] = zero8;

    constexpr int NT = D_ / 32;
    load_tiles(0);
    store_tiles(0);

    for (int t = 0; t < NT; ++t) {
        __syncthreads();                       // LDS buffer t&1 ready
        if (t + 1 < NT) load_tiles((t + 1) * 32);   // in flight during WMMA

        const unsigned int* AsU = (const unsigned int*)(As + (t & 1) * TILE_SH);
        const unsigned int* BsU = (const unsigned int*)(Bs + (t & 1) * TILE_SH);
        FragBF a[2], b[4];
#pragma unroll
        for (int f = 0; f < 2; ++f)
            load_frag_a(a[f], AsU, wm * 32 + f * 16 + lr, 0, 16, half);
#pragma unroll
        for (int f = 0; f < 4; ++f)
            load_frag_b(b[f], BsU, wn * 64 + f * 16 + lr, 0, 16, half);
#pragma unroll
        for (int fm = 0; fm < 2; ++fm)
#pragma unroll
            for (int fn = 0; fn < 4; ++fn)
                acc[fm][fn] = __builtin_amdgcn_wmma_f32_16x16x32_bf16(
                    false, a[fm].v, false, b[fn].v, (short)0, acc[fm][fn], false, false);

        if (t + 1 < NT) store_tiles((t + 1) & 1);   // waits loadcnt here
    }

    // Epilogue: bias (+ optional elu()+1), store bf16 or f32.
#pragma unroll
    for (int fm = 0; fm < 2; ++fm)
#pragma unroll
        for (int fn = 0; fn < 4; ++fn)
#pragma unroll
            for (int r = 0; r < 8; ++r) {
                int row = m0 + wm * 32 + fm * 16 + r + 8 * half;
                int col = n0 + wn * 64 + fn * 16 + lr;
                float v = acc[fm][fn][r] + bias[col];
                if (act) v = (v > 0.f) ? (v + 1.f) : __expf(v);  // elu(v)+1
                if (obf) obf[row * D_ + col] = f2bf(v);
                else     of32[row * D_ + col] = v;
            }
}

__global__ __launch_bounds__(256)
void gemm_qkv_kernel(const unsigned short* __restrict__ xb,
                     const unsigned short* __restrict__ wq,
                     const unsigned short* __restrict__ wk,
                     const unsigned short* __restrict__ wv,
                     const float* __restrict__ bq, const float* __restrict__ bk,
                     const float* __restrict__ bv,
                     unsigned short* __restrict__ qb,
                     unsigned short* __restrict__ kb,
                     unsigned short* __restrict__ vb) {
    __shared__ unsigned short As[2 * TILE_SH];
    __shared__ unsigned short Bs[2 * TILE_SH];
    int z = blockIdx.z;
    const unsigned short* W = (z == 0) ? wq : ((z == 1) ? wk : wv);
    const float* bias       = (z == 0) ? bq : ((z == 1) ? bk : bv);
    unsigned short* O       = (z == 0) ? qb : ((z == 1) ? kb : vb);
    gemm_body(xb, W, bias, O, nullptr, z < 2, As, Bs);
}

__global__ __launch_bounds__(256)
void gemm_proj_kernel(const unsigned short* __restrict__ A,
                      const unsigned short* __restrict__ W,
                      const float* __restrict__ bias,
                      float* __restrict__ Out) {
    __shared__ unsigned short As[2 * TILE_SH];
    __shared__ unsigned short Bs[2 * TILE_SH];
    gemm_body(A, W, bias, nullptr, Out, false, As, Bs);
}

// ---------------------------------------------------------------------------
// Chunked linear-attention scan. One block per (b,h), 4 waves (128 thr).
// Chunk C = 64 = hd; per chunk (all 64x64x64 WMMA GEMMs):
//   Scores  = tril(Q Kt)           (mask incl. diagonal)
//   Out     = Q @ T_prev + Scores @ V,  denom = q.z_prev + rowsum(Scores)
//   T      += Vt @ K  (T = S^T, kept f32 in LDS), z += colsum(K)
// ---------------------------------------------------------------------------
__global__ __launch_bounds__(128)
void linattn_scan_kernel(const unsigned short* __restrict__ Qg,
                         const unsigned short* __restrict__ Kg,
                         const unsigned short* __restrict__ Vg,
                         unsigned short* __restrict__ Og) {
    __shared__ float          Tst[64 * 64];   // state T = S^T, [m][d], f32
    __shared__ unsigned short Tb [64 * 64];   // bf16 copy of T for WMMA B-op
    __shared__ unsigned short Qs [64 * 64];   // Q chunk  [i][d]
    __shared__ unsigned short Ks [64 * 64];   // K chunk  [j][d]
    __shared__ unsigned short Kt [64 * 64];   // K^T      [d][j]
    __shared__ unsigned short Vt [64 * 64];   // V^T      [m][j]
    __shared__ float          Am [64 * 64];   // masked scores f32 [i][j]
    __shared__ unsigned short Ab [64 * 64];   // masked scores bf16
    __shared__ float          zvld[64];
    __shared__ float          qzd [64];

    const int tid  = threadIdx.x;
    const int bh   = blockIdx.x;
    const int b    = bh >> 4, h = bh & 15;
    const int wave = tid >> 5;
    const int wm   = wave >> 1, wn = wave & 1;
    const int lane = tid & 31, half = lane >> 4, lr = lane & 15;

    for (int i = tid; i < 4096; i += 128) { Tst[i] = 0.f; Tb[i] = 0; }
    if (tid < 64) zvld[tid] = 0.f;
    __syncthreads();

    const unsigned int* QsU = (const unsigned int*)Qs;
    const unsigned int* KsU = (const unsigned int*)Ks;
    const unsigned int* KtU = (const unsigned int*)Kt;
    const unsigned int* VtU = (const unsigned int*)Vt;
    const unsigned int* TbU = (const unsigned int*)Tb;
    const unsigned int* AbU = (const unsigned int*)Ab;

    const v8f zero8 = {0.f, 0.f, 0.f, 0.f, 0.f, 0.f, 0.f, 0.f};

    for (int c0 = 0; c0 < S_; c0 += 64) {
        // ---- stage chunk into LDS (2 threads / row, 32 bf16 each) ----
        {
            int j  = tid >> 1;
            int db = (tid & 1) * 32;
            long base = ((long)((b * S_ + c0 + j) * H_) + h) * HD_ + db;
#pragma unroll
            for (int u2 = 0; u2 < 4; ++u2) {
                uint4 qv = *(const uint4*)(Qg + base + u2 * 8);
                *(uint4*)(Qs + j * 64 + db + u2 * 8) = qv;
                uint4 kv = *(const uint4*)(Kg + base + u2 * 8);
                *(uint4*)(Ks + j * 64 + db + u2 * 8) = kv;
                uint4 vv = *(const uint4*)(Vg + base + u2 * 8);
                unsigned int kp[4] = {kv.x, kv.y, kv.z, kv.w};
                unsigned int vp[4] = {vv.x, vv.y, vv.z, vv.w};
#pragma unroll
                for (int p = 0; p < 4; ++p) {
                    int d0 = db + u2 * 8 + 2 * p;
                    Kt[(d0)     * 64 + j] = (unsigned short)(kp[p] & 0xffffu);
                    Kt[(d0 + 1) * 64 + j] = (unsigned short)(kp[p] >> 16);
                    Vt[(d0)     * 64 + j] = (unsigned short)(vp[p] & 0xffffu);
                    Vt[(d0 + 1) * 64 + j] = (unsigned short)(vp[p] >> 16);
                }
            }
        }
        __syncthreads();

        // ---- scores = Q @ K^T  (A=Qs [i][d], B=Ks [j][d]) ----
        v8f sA[2][2];
        sA[0][0] = zero8; sA[0][1] = zero8; sA[1][0] = zero8; sA[1][1] = zero8;
        mma64(QsU, KsU, wm, wn, half, lr, sA);

        // causal mask (j <= i), stash f32 scores
#pragma unroll
        for (int fm = 0; fm < 2; ++fm)
#pragma unroll
            for (int fn = 0; fn < 2; ++fn)
#pragma unroll
                for (int r = 0; r < 8; ++r) {
                    int i  = wm * 32 + fm * 16 + r + 8 * half;
                    int jj = wn * 32 + fn * 16 + lr;
                    Am[i * 64 + jj] = (jj <= i) ? sA[fm][fn][r] : 0.f;
                }
        __syncthreads();

        // ---- denominators + bf16 scores ----
        if (tid < 64) {
            int i = tid;
            float rs = 0.f;
            for (int jj = 0; jj < 64; ++jj) rs += Am[i * 64 + jj];
            float qd = 0.f;
            for (int d = 0; d < 64; ++d) qd += bf2f(Qs[i * 64 + d]) * zvld[d];
            qzd[i] = rs + qd + 1e-6f;
        } else {
            int i = tid - 64;
            for (int jj = 0; jj < 64; ++jj) Ab[i * 64 + jj] = f2bf(Am[i * 64 + jj]);
        }
        __syncthreads();

        // ---- out = Q @ T_prev + Scores @ V ----
        v8f oA[2][2];
        oA[0][0] = zero8; oA[0][1] = zero8; oA[1][0] = zero8; oA[1][1] = zero8;
        mma64(QsU, TbU, wm, wn, half, lr, oA);   // inter-chunk
        mma64(AbU, VtU, wm, wn, half, lr, oA);   // intra-chunk
#pragma unroll
        for (int fm = 0; fm < 2; ++fm)
#pragma unroll
            for (int fn = 0; fn < 2; ++fn)
#pragma unroll
                for (int r = 0; r < 8; ++r) {
                    int i = wm * 32 + fm * 16 + r + 8 * half;
                    int m = wn * 32 + fn * 16 + lr;
                    float val = oA[fm][fn][r] / qzd[i];
                    Og[(long)((b * S_ + c0 + i)) * D_ + h * HD_ + m] = f2bf(val);
                }

        // ---- state update: T += V^T @ K  (A=Vt [m][j], B=Kt [d][j]) ----
        v8f tD[2][2];
        tD[0][0] = zero8; tD[0][1] = zero8; tD[1][0] = zero8; tD[1][1] = zero8;
        mma64(VtU, KtU, wm, wn, half, lr, tD);
#pragma unroll
        for (int fm = 0; fm < 2; ++fm)
#pragma unroll
            for (int fn = 0; fn < 2; ++fn)
#pragma unroll
                for (int r = 0; r < 8; ++r) {
                    int m = wm * 32 + fm * 16 + r + 8 * half;
                    int d = wn * 32 + fn * 16 + lr;
                    Tst[m * 64 + d] += tD[fm][fn][r];
                }
        __syncthreads();

        // ---- z += colsum(K); refresh bf16 state copy ----
        if (tid < 64) {
            int d = tid;
            float s = 0.f;
            for (int j = 0; j < 64; ++j) s += bf2f(Kt[d * 64 + j]);
            zvld[d] += s;
        }
        for (int i2 = tid; i2 < 4096; i2 += 128) Tb[i2] = f2bf(Tst[i2]);
        __syncthreads();
    }
}

// ---------------------------------------------------------------------------
// Host-side launch
// ---------------------------------------------------------------------------
extern "C" void kernel_launch(void* const* d_in, const int* in_sizes, int n_in,
                              void* d_out, int out_size, void* d_ws, size_t ws_size,
                              hipStream_t stream) {
    (void)in_sizes; (void)n_in; (void)out_size; (void)ws_size;

    const float* x  = (const float*)d_in[0];
    // d_in[1] = mask (unused, as in reference)
    const float* Wq = (const float*)d_in[2];
    const float* bq = (const float*)d_in[3];
    const float* Wk = (const float*)d_in[4];
    const float* bk = (const float*)d_in[5];
    const float* Wv = (const float*)d_in[6];
    const float* bv = (const float*)d_in[7];
    const float* Wo = (const float*)d_in[8];
    const float* bo = (const float*)d_in[9];
    float* out = (float*)d_out;

    char* ws = (char*)d_ws;
    const size_t szXB = (size_t)MTOT * D_ * sizeof(unsigned short);  // 16 MiB
    const size_t szW  = (size_t)D_ * D_ * sizeof(unsigned short);    //  2 MiB

    unsigned short* xb    = (unsigned short*)(ws);
    unsigned short* wqb   = (unsigned short*)(ws + szXB);
    unsigned short* wkb   = (unsigned short*)(ws + szXB + 1 * szW);
    unsigned short* wvb   = (unsigned short*)(ws + szXB + 2 * szW);
    unsigned short* wob   = (unsigned short*)(ws + szXB + 3 * szW);
    unsigned short* qb    = (unsigned short*)(ws + szXB + 4 * szW);
    unsigned short* kb    = qb + (size_t)MTOT * D_;
    unsigned short* vb    = kb + (size_t)MTOT * D_;
    unsigned short* valsb = vb + (size_t)MTOT * D_;

    // 1) f32 -> bf16 conversions
    const int nX = MTOT * D_;     // 8388608
    const int nW = D_ * D_;       // 1048576
    cvt_f32_bf16_kernel<<<(nX / 2 + 255) / 256, 256, 0, stream>>>(x,  xb,  nX);
    cvt_f32_bf16_kernel<<<(nW / 2 + 255) / 256, 256, 0, stream>>>(Wq, wqb, nW);
    cvt_f32_bf16_kernel<<<(nW / 2 + 255) / 256, 256, 0, stream>>>(Wk, wkb, nW);
    cvt_f32_bf16_kernel<<<(nW / 2 + 255) / 256, 256, 0, stream>>>(Wv, wvb, nW);
    cvt_f32_bf16_kernel<<<(nW / 2 + 255) / 256, 256, 0, stream>>>(Wo, wob, nW);

    // 2) fused QKV projections (+ elu+1 on Q,K)
    gemm_qkv_kernel<<<dim3(D_ / 128, MTOT / 128, 3), 256, 0, stream>>>(
        xb, wqb, wkb, wvb, bq, bk, bv, qb, kb, vb);

    // 3) chunked linear-attention scan (one block per (b,h))
    linattn_scan_kernel<<<B_ * H_, 128, 0, stream>>>(qb, kb, vb, valsb);

    // 4) output projection
    gemm_proj_kernel<<<dim3(D_ / 128, MTOT / 128), 256, 0, stream>>>(
        valsb, wob, bo, out);
}
